// SwinTransformerLayer_53601191854233
// MI455X (gfx1250) — matrix operand order repaired and checked by reading, not compiled, not run on hardware
//
#include <hip/hip_runtime.h>
#include <hip/hip_bf16.h>
#include <math.h>

// ---------------------------------------------------------------------------
// Swin block on gfx1250: all GEMMs via v_wmma_f32_16x16x32_f16 (wave32 WMMA),
// global->LDS staging via GLOBAL_LOAD_ASYNC_TO_LDS_B128 (ASYNCcnt-tracked),
// double-buffered 128x64 tiles, 8 waves per workgroup.
// ---------------------------------------------------------------------------

typedef __attribute__((ext_vector_type(16))) _Float16 v16h;
typedef __attribute__((ext_vector_type(8)))  _Float16 v8h;
typedef __attribute__((ext_vector_type(8)))  float    v8f;

#define DIM   192
#define HEADS 6
#define HD    32
#define NTOK  49
#define HID   768
#define MROWS 50176          // 16 * 3136 tokens (= 1024 windows * 49)
#define SCALE 0.17677669529663687f

#if __has_builtin(__builtin_amdgcn_global_load_async_to_lds_b128)
#define USE_ASYNC_LDS 1
#else
#define USE_ASYNC_LDS 0
#endif

// b128 payload type as the builtin expects: GCC-vector of 4 ints.
typedef int v4i_b128 __attribute__((vector_size(16)));
typedef __attribute__((address_space(1))) v4i_b128* gptr_b128;
typedef __attribute__((address_space(3))) v4i_b128* lptr_b128;

#define AS1P(p) ((gptr_b128)(const_cast<_Float16*>(p)))
#define AS3P(p) ((lptr_b128)(p))

union FragU { v16h v; v8h h2[2]; };

static __device__ __forceinline__ v8f wmma_f16(v16h a, v16h b, v8f c) {
    return __builtin_amdgcn_wmma_f32_16x16x32_f16(false, a, false, b, (short)0, c, false, false);
}

static __device__ __forceinline__ void wait_async_lds() {
#if USE_ASYNC_LDS
#if __has_builtin(__builtin_amdgcn_s_wait_asynccnt)
    __builtin_amdgcn_s_wait_asynccnt(0);
#else
    asm volatile("s_wait_asynccnt 0x0" ::: "memory");
#endif
#endif
}

// Stage this thread's slice of one k-step tile: 16 halfs of A (2 x b128) and
// 8 halfs of B (1 x b128), either async direct-to-LDS or via VGPRs.
static __device__ __forceinline__ void stage_tile(
    const _Float16* __restrict__ gA, const _Float16* __restrict__ gB,
    _Float16* aDst, _Float16* bDst)
{
#if USE_ASYNC_LDS
    __builtin_amdgcn_global_load_async_to_lds_b128(AS1P(gA),     AS3P(aDst),     0, 0);
    __builtin_amdgcn_global_load_async_to_lds_b128(AS1P(gA + 8), AS3P(aDst + 8), 0, 0);
    __builtin_amdgcn_global_load_async_to_lds_b128(AS1P(gB),     AS3P(bDst),     0, 0);
#else
    v8h a0 = *(const v8h*)gA;
    v8h a1 = *(const v8h*)(gA + 8);
    v8h b0 = *(const v8h*)gB;
    *(v8h*)aDst       = a0;
    *(v8h*)(aDst + 8) = a1;
    *(v8h*)bDst       = b0;
#endif
}

// ---------------------------------------------------------------------------
// GEMM core: C_tile(128x64) = A(M x K, f16 rm) @ B(N x K, f16 rm)^T
// 256 threads = 8 waves; wave wv computes rows [wv*16, wv*16+16) of the tile.
// LDS double buffered: As = 2 x 128x32, Bs = 2 x 64x32.
// Fragment layouts used here (contiguous k, same order for A and B):
//   A 16x32 f16: lane m=l&15, hi=l>>4; halves e  <-> K=hi*16+e
//   B 32x16 f16: lane n=l&15, hi=l>>4; halves e  <-> K=hi*16+e (row n of the NxK matrix)
// ---------------------------------------------------------------------------
static __device__ __forceinline__ void gemm_tile_core(
    const _Float16* __restrict__ A, const _Float16* __restrict__ B,
    int K, int m0, int n0, _Float16* As, _Float16* Bs, v8f acc[4])
{
    const int tid  = threadIdx.x;
    const int wv   = tid >> 5;
    const int lane = tid & 31;
    const int hi   = lane >> 4;
    const int ln   = lane & 15;
    for (int j = 0; j < 4; ++j)
        for (int e = 0; e < 8; ++e) acc[j][e] = 0.0f;

    const int ra = tid >> 1;          // 0..127 A-tile row
    const int ha = (tid & 1) * 16;    // 16-half chunk within 32-wide k-slice
    const int rb = tid >> 2;          // 0..63  B-tile row
    const int qb = (tid & 3) * 8;     // 8-half chunk

    const _Float16* gA = A + (size_t)(m0 + ra) * K + ha;
    const _Float16* gB = B + (size_t)(n0 + rb) * K + qb;
    const int aoff = ra * 32 + ha;
    const int boff = rb * 32 + qb;
    const int nsteps = K >> 5;

    stage_tile(gA, gB, As + aoff, Bs + boff);           // tile 0 -> buffer 0

    for (int s = 0; s < nsteps; ++s) {
        wait_async_lds();
        __syncthreads();
        const int p = s & 1;
        if (s + 1 < nsteps) {
            const int q = (s + 1) & 1;
            stage_tile(gA + (s + 1) * 32, gB + (s + 1) * 32,
                       As + q * 4096 + aoff, Bs + q * 2048 + boff);
        }
        FragU ua;
        const _Float16* ap = As + p * 4096 + (wv * 16 + ln) * 32;
        ua.h2[0] = *(const v8h*)(ap + hi * 16);
        ua.h2[1] = *(const v8h*)(ap + hi * 16 + 8);
        const _Float16* bbase = Bs + p * 2048;
        #pragma unroll
        for (int j = 0; j < 4; ++j) {
            FragU ub;
            const _Float16* bp = bbase + (j * 16 + ln) * 32 + hi * 16;
            ub.h2[0] = *(const v8h*)bp;
            ub.h2[1] = *(const v8h*)(bp + 8);
            acc[j] = wmma_f16(ua.v, ub.v, acc[j]);
        }
    }
}

// ---------------------------------------------------------------------------
// f32 -> f16 weight conversion
// ---------------------------------------------------------------------------
__global__ void cvt_f16_kernel(const float* __restrict__ in, _Float16* __restrict__ out, int n) {
    int i = blockIdx.x * blockDim.x + threadIdx.x;
    if (i < n) out[i] = (_Float16)in[i];
}

// ---------------------------------------------------------------------------
// LN1 + roll(-3,-3) + window partition. One thread per output row m.
// ---------------------------------------------------------------------------
__global__ void ln1_window_kernel(const float* __restrict__ x,
                                  const float* __restrict__ g,
                                  const float* __restrict__ bb,
                                  _Float16* __restrict__ xw)
{
    int m = blockIdx.x * blockDim.x + threadIdx.x;
    if (m >= MROWS) return;
    int b   = m / 3136, rem = m % 3136;
    int wi  = rem / NTOK, i = rem % NTOK;
    int wh  = wi >> 3, wwc = wi & 7;
    int r   = i / 7, c = i % 7;
    int hsrc = wh * 7 + r + 3;  if (hsrc >= 56) hsrc -= 56;
    int wsrc = wwc * 7 + c + 3; if (wsrc >= 56) wsrc -= 56;
    const float* xp = x + (size_t)b * DIM * 3136 + hsrc * 56 + wsrc;
    float s = 0.f, s2 = 0.f;
    for (int cc = 0; cc < DIM; ++cc) { float v = xp[(size_t)cc * 3136]; s += v; s2 += v * v; }
    float mu = s * (1.f / DIM);
    float var = s2 * (1.f / DIM) - mu * mu;
    float rs = rsqrtf(var + 1e-5f);
    _Float16* op = xw + (size_t)m * DIM;
    for (int cc = 0; cc < DIM; ++cc) {
        float v = xp[(size_t)cc * 3136];
        op[cc] = (_Float16)((v - mu) * rs * g[cc] + bb[cc]);
    }
}

// ---------------------------------------------------------------------------
// QKV GEMM: (50176x192) @ (576x192)^T + bias -> qkv f16 (50176x576)
// ---------------------------------------------------------------------------
__global__ void gemm_qkv_kernel(const _Float16* __restrict__ A, const _Float16* __restrict__ B,
                                const float* __restrict__ bias, _Float16* __restrict__ C)
{
    __shared__ _Float16 As[2 * 128 * 32], Bs[2 * 64 * 32];
    v8f acc[4];
    int m0 = blockIdx.x * 128, n0 = blockIdx.y * 64;
    gemm_tile_core(A, B, DIM, m0, n0, As, Bs, acc);
    int tid = threadIdx.x, wv = tid >> 5, lane = tid & 31, hi = lane >> 4, ln = lane & 15;
    #pragma unroll
    for (int j = 0; j < 4; ++j) {
        int colg = n0 + j * 16 + ln;
        float bi = bias[colg];
        #pragma unroll
        for (int rr = 0; rr < 8; ++rr) {
            int rowg = m0 + wv * 16 + hi * 8 + rr;
            C[(size_t)rowg * 576 + colg] = (_Float16)(acc[j][rr] + bi);
        }
    }
}

// ---------------------------------------------------------------------------
// Windowed attention: one workgroup per (window, head). 128 threads = 4 waves.
// S = (Q*scale) Kt via WMMA; softmax (+rpb +shift-mask) in f32; O = P V via WMMA.
// ---------------------------------------------------------------------------
__global__ void attn_kernel(const _Float16* __restrict__ qkv,
                            const float* __restrict__ rpb,
                            _Float16* __restrict__ outb)
{
    __shared__ _Float16 qs[64 * 32];
    __shared__ _Float16 ks[64 * 32];
    __shared__ _Float16 vT[32 * 64];
    __shared__ float    sb[64 * 64];
    __shared__ _Float16 ps[64 * 64];

    int bid = blockIdx.x;
    int w = bid / HEADS, h = bid % HEADS;
    int tid = threadIdx.x;
    int wi = w & 63, wh = wi >> 3, wwc = wi & 7;

    for (int idx = tid; idx < 2048; idx += 128) {
        int n = idx >> 5, d = idx & 31;
        _Float16 qv = (_Float16)0.f, kv = (_Float16)0.f;
        if (n < NTOK) {
            size_t base = (size_t)(w * NTOK + n) * 576 + h * HD + d;
            qv = (_Float16)((float)qkv[base] * SCALE);
            kv = qkv[base + DIM];
        }
        qs[n * 32 + d] = qv;
        ks[n * 32 + d] = kv;
    }
    for (int idx = tid; idx < 2048; idx += 128) {
        int n = idx >> 5, d = idx & 31;
        _Float16 vv = (_Float16)0.f;
        if (n < NTOK) vv = qkv[(size_t)(w * NTOK + n) * 576 + 2 * DIM + h * HD + d];
        vT[d * 64 + n] = vv;
    }
    __syncthreads();

    int wv = tid >> 5, lane = tid & 31, hi = lane >> 4, ln = lane & 15;

    // S strip (16 rows x 64 cols) per wave
    {
        FragU ua;
        const _Float16* ap = qs + (wv * 16 + ln) * 32;
        ua.h2[0] = *(const v8h*)(ap + hi * 16);
        ua.h2[1] = *(const v8h*)(ap + hi * 16 + 8);
        #pragma unroll
        for (int j = 0; j < 4; ++j) {
            FragU ub;
            const _Float16* bp = ks + (j * 16 + ln) * 32 + hi * 16;
            ub.h2[0] = *(const v8h*)bp;
            ub.h2[1] = *(const v8h*)(bp + 8);
            v8f acc; for (int e = 0; e < 8; ++e) acc[e] = 0.f;
            acc = wmma_f16(ua.v, ub.v, acc);
            #pragma unroll
            for (int rr = 0; rr < 8; ++rr)
                sb[(wv * 16 + hi * 8 + rr) * 64 + j * 16 + ln] = acc[rr];
        }
    }
    __syncthreads();

    // softmax rows (only i<49 meaningful); rpb + shift mask computed inline
    if (tid < 64) {
        int i = tid;
        if (i < NTOK) {
            int r1 = i / 7, c1 = i % 7;
            int hs1 = wh * 7 + r1, ws1 = wwc * 7 + c1;
            int id1 = (hs1 < 49 ? 0 : (hs1 < 53 ? 1 : 2)) * 3 + (ws1 < 49 ? 0 : (ws1 < 53 ? 1 : 2));
            float mx = -1e30f;
            for (int j = 0; j < NTOK; ++j) {
                int r2 = j / 7, c2 = j % 7;
                int hs2 = wh * 7 + r2, ws2 = wwc * 7 + c2;
                int id2 = (hs2 < 49 ? 0 : (hs2 < 53 ? 1 : 2)) * 3 + (ws2 < 49 ? 0 : (ws2 < 53 ? 1 : 2));
                int rel = (r1 - r2 + 6) * 13 + (c1 - c2 + 6);
                float v = sb[i * 64 + j] + rpb[rel * HEADS + h] + (id1 == id2 ? 0.f : -100.f);
                sb[i * 64 + j] = v;
                mx = fmaxf(mx, v);
            }
            float sum = 0.f;
            for (int j = 0; j < NTOK; ++j) {
                float e = __expf(sb[i * 64 + j] - mx);
                sb[i * 64 + j] = e;
                sum += e;
            }
            float inv = 1.f / sum;
            for (int j = 0; j < NTOK; ++j) ps[i * 64 + j] = (_Float16)(sb[i * 64 + j] * inv);
            for (int j = NTOK; j < 64; ++j) ps[i * 64 + j] = (_Float16)0.f;
        } else {
            for (int j = 0; j < 64; ++j) ps[i * 64 + j] = (_Float16)0.f;
        }
    }
    __syncthreads();

    // O = P @ V : per-wave 16x32 strip, K=64 in two 32-steps
    v8f oacc[2];
    for (int j = 0; j < 2; ++j) for (int e = 0; e < 8; ++e) oacc[j][e] = 0.f;
    #pragma unroll
    for (int kstep = 0; kstep < 2; ++kstep) {
        FragU ua;
        const _Float16* ap = ps + (wv * 16 + ln) * 64 + kstep * 32;
        ua.h2[0] = *(const v8h*)(ap + hi * 16);
        ua.h2[1] = *(const v8h*)(ap + hi * 16 + 8);
        #pragma unroll
        for (int j = 0; j < 2; ++j) {
            FragU ub;
            const _Float16* bp = vT + (j * 16 + ln) * 64 + kstep * 32 + hi * 16;
            ub.h2[0] = *(const v8h*)bp;
            ub.h2[1] = *(const v8h*)(bp + 8);
            oacc[j] = wmma_f16(ua.v, ub.v, oacc[j]);
        }
    }
    #pragma unroll
    for (int j = 0; j < 2; ++j)
        for (int rr = 0; rr < 8; ++rr) {
            int i = wv * 16 + hi * 8 + rr;
            if (i < NTOK)
                outb[(size_t)(w * NTOK + i) * DIM + h * HD + j * 16 + ln] = (_Float16)oacc[j][rr];
        }
}

// ---------------------------------------------------------------------------
// proj GEMM fused with reverse window + roll(+3,+3) + residual:
// x2[b,h,w,c] = x[b,c,h,w] + (attnout @ proj_w^T + proj_b) mapped back
// ---------------------------------------------------------------------------
__global__ void gemm_proj_kernel(const _Float16* __restrict__ A, const _Float16* __restrict__ B,
                                 const float* __restrict__ bias,
                                 const float* __restrict__ x, float* __restrict__ x2)
{
    __shared__ _Float16 As[2 * 128 * 32], Bs[2 * 64 * 32];
    v8f acc[4];
    int m0 = blockIdx.x * 128, n0 = blockIdx.y * 64;
    gemm_tile_core(A, B, DIM, m0, n0, As, Bs, acc);
    int tid = threadIdx.x, wv = tid >> 5, lane = tid & 31, hi = lane >> 4, ln = lane & 15;
    #pragma unroll
    for (int j = 0; j < 4; ++j) {
        int colg = n0 + j * 16 + ln;
        float bi = bias[colg];
        #pragma unroll
        for (int rr = 0; rr < 8; ++rr) {
            int m = m0 + wv * 16 + hi * 8 + rr;
            int b = m / 3136, rem = m % 3136;
            int wi = rem / NTOK, i = rem % NTOK;
            int wh = wi >> 3, wwc = wi & 7;
            int r = i / 7, c = i % 7;
            int hh = wh * 7 + r + 3;  if (hh >= 56) hh -= 56;   // roll back +3
            int ww = wwc * 7 + c + 3; if (ww >= 56) ww -= 56;
            float val = acc[j][rr] + bi + x[((size_t)(b * DIM + colg) * 56 + hh) * 56 + ww];
            x2[(size_t)(b * 3136 + hh * 56 + ww) * DIM + colg] = val;
        }
    }
}

// ---------------------------------------------------------------------------
// LN2 on x2 token rows -> f16
// ---------------------------------------------------------------------------
__global__ void ln2_kernel(const float* __restrict__ x2, const float* __restrict__ g,
                           const float* __restrict__ bb, _Float16* __restrict__ out)
{
    int m = blockIdx.x * blockDim.x + threadIdx.x;
    if (m >= MROWS) return;
    const float* xp = x2 + (size_t)m * DIM;
    float s = 0.f, s2 = 0.f;
    for (int cc = 0; cc < DIM; ++cc) { float v = xp[cc]; s += v; s2 += v * v; }
    float mu = s * (1.f / DIM);
    float var = s2 * (1.f / DIM) - mu * mu;
    float rs = rsqrtf(var + 1e-5f);
    _Float16* op = out + (size_t)m * DIM;
    for (int cc = 0; cc < DIM; ++cc)
        op[cc] = (_Float16)((xp[cc] - mu) * rs * g[cc] + bb[cc]);
}

// ---------------------------------------------------------------------------
// FC1 GEMM + exact GELU -> f16 (50176 x 768)
// ---------------------------------------------------------------------------
__global__ void gemm_fc1_kernel(const _Float16* __restrict__ A, const _Float16* __restrict__ B,
                                const float* __restrict__ bias, _Float16* __restrict__ C)
{
    __shared__ _Float16 As[2 * 128 * 32], Bs[2 * 64 * 32];
    v8f acc[4];
    int m0 = blockIdx.x * 128, n0 = blockIdx.y * 64;
    gemm_tile_core(A, B, DIM, m0, n0, As, Bs, acc);
    int tid = threadIdx.x, wv = tid >> 5, lane = tid & 31, hi = lane >> 4, ln = lane & 15;
    #pragma unroll
    for (int j = 0; j < 4; ++j) {
        int colg = n0 + j * 16 + ln;
        float bi = bias[colg];
        #pragma unroll
        for (int rr = 0; rr < 8; ++rr) {
            int rowg = m0 + wv * 16 + hi * 8 + rr;
            float v = acc[j][rr] + bi;
            float gl = 0.5f * v * (1.0f + erff(v * 0.7071067811865475f));
            C[(size_t)rowg * HID + colg] = (_Float16)gl;
        }
    }
}

// ---------------------------------------------------------------------------
// FC2 GEMM + residual + NHWC->NCHW scatter into d_out
// ---------------------------------------------------------------------------
__global__ void gemm_fc2_kernel(const _Float16* __restrict__ A, const _Float16* __restrict__ B,
                                const float* __restrict__ bias,
                                const float* __restrict__ x2, float* __restrict__ outp)
{
    __shared__ _Float16 As[2 * 128 * 32], Bs[2 * 64 * 32];
    v8f acc[4];
    int m0 = blockIdx.x * 128, n0 = blockIdx.y * 64;
    gemm_tile_core(A, B, HID, m0, n0, As, Bs, acc);
    int tid = threadIdx.x, wv = tid >> 5, lane = tid & 31, hi = lane >> 4, ln = lane & 15;
    #pragma unroll
    for (int j = 0; j < 4; ++j) {
        int colg = n0 + j * 16 + ln;
        float bi = bias[colg];
        #pragma unroll
        for (int rr = 0; rr < 8; ++rr) {
            int rowg = m0 + wv * 16 + hi * 8 + rr;
            float val = acc[j][rr] + bi + x2[(size_t)rowg * DIM + colg];
            int b = rowg / 3136, rem = rowg % 3136;
            int hh = rem / 56, ww = rem % 56;
            outp[((size_t)(b * DIM + colg) * 56 + hh) * 56 + ww] = val;
        }
    }
}

// ---------------------------------------------------------------------------
extern "C" void kernel_launch(void* const* d_in, const int* in_sizes, int n_in,
                              void* d_out, int out_size, void* d_ws, size_t ws_size,
                              hipStream_t stream) {
    const float* x       = (const float*)d_in[0];
    const float* qkv_w   = (const float*)d_in[1];
    const float* qkv_b   = (const float*)d_in[2];
    const float* proj_w  = (const float*)d_in[3];
    const float* proj_b  = (const float*)d_in[4];
    const float* rpb_tab = (const float*)d_in[5];
    const float* n1g     = (const float*)d_in[6];
    const float* n1b     = (const float*)d_in[7];
    const float* n2g     = (const float*)d_in[8];
    const float* n2b     = (const float*)d_in[9];
    const float* fc1_w   = (const float*)d_in[10];
    const float* fc1_b   = (const float*)d_in[11];
    const float* fc2_w   = (const float*)d_in[12];
    const float* fc2_b   = (const float*)d_in[13];
    float* outp = (float*)d_out;

    char* ws = (char*)d_ws;
    size_t off = 0;
    auto carve = [&](size_t bytes) { size_t o = off; off += (bytes + 255) & ~(size_t)255; return o; };
    _Float16* wq     = (_Float16*)(ws + carve((size_t)576 * 192 * 2));
    _Float16* wp     = (_Float16*)(ws + carve((size_t)192 * 192 * 2));
    _Float16* w1     = (_Float16*)(ws + carve((size_t)768 * 192 * 2));
    _Float16* w2     = (_Float16*)(ws + carve((size_t)192 * 768 * 2));
    char*     regB   = ws + carve((size_t)MROWS * DIM * 2);     // xw, later attnout
    char*     regC   = ws + carve((size_t)MROWS * 576 * 2);     // qkv, later ln2 out
    float*    x2     = (float*)(ws + carve((size_t)MROWS * DIM * 4));
    _Float16* fc1out = (_Float16*)(ws + carve((size_t)MROWS * HID * 2));

    _Float16* xw      = (_Float16*)regB;
    _Float16* attnout = (_Float16*)regB;   // reuses xw after QKV GEMM consumed it
    _Float16* qkvbuf  = (_Float16*)regC;
    _Float16* ln2out  = (_Float16*)regC;   // reuses qkv after attention consumed it

    cvt_f16_kernel<<<(576 * 192 + 255) / 256, 256, 0, stream>>>(qkv_w, wq, 576 * 192);
    cvt_f16_kernel<<<(192 * 192 + 255) / 256, 256, 0, stream>>>(proj_w, wp, 192 * 192);
    cvt_f16_kernel<<<(768 * 192 + 255) / 256, 256, 0, stream>>>(fc1_w, w1, 768 * 192);
    cvt_f16_kernel<<<(192 * 768 + 255) / 256, 256, 0, stream>>>(fc2_w, w2, 192 * 768);

    ln1_window_kernel<<<(MROWS + 255) / 256, 256, 0, stream>>>(x, n1g, n1b, xw);

    gemm_qkv_kernel<<<dim3(MROWS / 128, 576 / 64), 256, 0, stream>>>(xw, wq, qkv_b, qkvbuf);

    attn_kernel<<<1024 * HEADS, 128, 0, stream>>>(qkvbuf, rpb_tab, attnout);

    gemm_proj_kernel<<<dim3(MROWS / 128, DIM / 64), 256, 0, stream>>>(attnout, wp, proj_b, x, x2);

    ln2_kernel<<<(MROWS + 255) / 256, 256, 0, stream>>>(x2, n2g, n2b, ln2out);

    gemm_fc1_kernel<<<dim3(MROWS / 128, HID / 64), 256, 0, stream>>>(ln2out, w1, fc1_b, fc1out);

    gemm_fc2_kernel<<<dim3(MROWS / 128, DIM / 64), 256, 0, stream>>>(fc1out, w2, fc2_b, x2, outp);

    (void)in_sizes; (void)n_in; (void)out_size; (void)ws_size;
}
